// AttentionBlock_2534030705303
// MI455X (gfx1250) — matrix operand order repaired and checked by reading, not compiled, not run on hardware
//
#include <hip/hip_runtime.h>
#include <hip/hip_bf16.h>
#include <stdint.h>

// Problem constants from the reference: B=4, D=128, T=4096, K=64, V=128.
#define B_DIM 4
#define D_DIM 128
#define T_DIM 4096
#define KP_DIM 64
#define V_DIM 128

#define NEG_BIG (-3.0e38f)

typedef __attribute__((ext_vector_type(16))) __bf16 v16bf;
typedef __attribute__((ext_vector_type(8)))  float  v8f;

__device__ __forceinline__ __bf16 f2bf(float f) {
  return static_cast<__bf16>(f);   // native v_cvt path on gfx1250
}

// A-fragment (16x32 bf16) load from row-major src[row][k], row = lane&15.
// Lanes 0-15 hold K {0..15}, lanes 16-31 hold K {16..31} (contiguous, matches B-frag).
__device__ __forceinline__ v16bf load_a_frag(const __bf16* base, int stride) {
  const int lane = threadIdx.x & 31;
  const int row = lane & 15, half = lane >> 4;
  const __bf16* p = base + row * stride;
  const int k0 = 16 * half, k1 = 16 * half + 8;
  v16bf a;
#pragma unroll
  for (int j = 0; j < 8; ++j) a[j] = p[k0 + j];
#pragma unroll
  for (int j = 0; j < 8; ++j) a[8 + j] = p[k1 + j];
  return a;
}

// B-fragment (32x16 bf16): lane holds column n = lane&15, K contiguous 16 at 16*(lane>>4).
// Source is "B transposed" row-major: srcT[n][k]  (K^T tiles come straight from Kp[t][k]).
__device__ __forceinline__ v16bf load_b_frag(const __bf16* srcT, int stride) {
  const int lane = threadIdx.x & 31;
  const int n = lane & 15, half = lane >> 4;
  const __bf16* p = srcT + n * stride + 16 * half;
  v16bf b;
#pragma unroll
  for (int j = 0; j < 16; ++j) b[j] = p[j];
  return b;
}

__device__ __forceinline__ v8f wmma_bf16(v16bf a, v16bf b, v8f c) {
  return __builtin_amdgcn_wmma_f32_16x16x32_bf16(false, a, false, b, (short)0, c,
                                                 false, false);
}

// ---------------------------------------------------------------------------
// Kernel 0: one-time weight conversion to bf16.
// Wb rows: [0,64) = Wq, [64,128) = Wk, [128,256) = Wv; each row 128 bf16.
// ---------------------------------------------------------------------------
__global__ __launch_bounds__(256) void wconv_kernel(
    const float* __restrict__ Wq, const float* __restrict__ Wk,
    const float* __restrict__ Wv, __bf16* __restrict__ Wb) {
  const int i = blockIdx.x * 256 + threadIdx.x;   // 32 blocks -> 8192 threads x4
#pragma unroll
  for (int j = 0; j < 4; ++j) {
    const int idx = i * 4 + j;                    // 0 .. 32767
    const int row = idx >> 7, col = idx & 127;
    float v;
    if (row < 64)        v = Wq[row * D_DIM + col];
    else if (row < 128)  v = Wk[(row - 64) * D_DIM + col];
    else                 v = Wv[(row - 128) * D_DIM + col];
    Wb[idx] = f2bf(v);
  }
}

// ---------------------------------------------------------------------------
// Kernel 1: projections. Qp,Kp bf16 [B,T,64] row-major; VpT bf16 [B,128,T].
// One block per (b, 64-row t-tile); x tile staged through LDS (transposing the
// [D,T] input to [t,d] so A-fragments are contiguous reads).
// ---------------------------------------------------------------------------
__global__ __launch_bounds__(256) void proj_kernel(
    const float* __restrict__ mb,
    const __bf16* __restrict__ Wb,
    const float* __restrict__ bk, const float* __restrict__ bq,
    const float* __restrict__ bv,
    __bf16* __restrict__ Qp, __bf16* __restrict__ Kp, __bf16* __restrict__ VpT) {
  __shared__ __bf16 xs[64 * 136];  // [t=64][d=128] padded stride 136
  const int tid = threadIdx.x;
  const int b = blockIdx.x >> 6;          // T/64 = 64 tiles per batch
  const int t0 = (blockIdx.x & 63) * 64;
  const float* xb = mb + (size_t)b * D_DIM * T_DIM;

  for (int idx = tid; idx < 64 * 128; idx += 256) {
    const int d = idx >> 6, tl = idx & 63;          // coalesced along t
    xs[tl * 136 + d] = f2bf(xb[(size_t)d * T_DIM + (t0 + tl)]);
  }
  __syncthreads();

  const int wave = tid >> 5, lane = tid & 31;
  const int lrow = lane & 15, lhalf = lane >> 4;

  for (int tile = wave; tile < 64; tile += 8) {
    const int ts = tile >> 4;   // t sub-tile 0..3
    const int ns = tile & 15;   // output column group
    // Wb row groups: ns<4 -> Wq rows [0,64); ns<8 -> Wk rows [64,128); else Wv rows [128,256)
    const float* bias; int wrow0, nc0, which;
    if (ns < 4)      { bias = bq; wrow0 = ns * 16;              nc0 = ns * 16;       which = 0; }
    else if (ns < 8) { bias = bk; wrow0 = 64 + (ns - 4) * 16;   nc0 = (ns - 4) * 16; which = 1; }
    else             { bias = bv; wrow0 = 128 + (ns - 8) * 16;  nc0 = (ns - 8) * 16; which = 2; }

    v8f c;
    const float bini = bias[nc0 + lrow];     // bias broadcast down the column
#pragma unroll
    for (int r = 0; r < 8; ++r) c[r] = bini;

#pragma unroll
    for (int kk = 0; kk < 4; ++kk) {         // D=128 -> 4 x K32
      v16bf a  = load_a_frag(xs + ts * 16 * 136 + kk * 32, 136);
      v16bf bb = load_b_frag(Wb + (size_t)wrow0 * D_DIM + kk * 32, D_DIM);
      c = wmma_bf16(a, bb, c);
    }

    if (which == 2) {  // V stored transposed: VpT[(b*128+v)*T + t]
      __bf16* o = VpT + ((size_t)(b * V_DIM + nc0 + lrow)) * T_DIM
                      + t0 + ts * 16 + 8 * lhalf;
#pragma unroll
      for (int r = 0; r < 8; ++r) o[r] = f2bf(c[r]);
    } else {
      __bf16* base = (which == 0) ? Qp : Kp;
      __bf16* o = base + ((size_t)(b * T_DIM + t0 + ts * 16 + 8 * lhalf)) * KP_DIM
                       + nc0 + lrow;
#pragma unroll
      for (int r = 0; r < 8; ++r) o[(size_t)r * KP_DIM] = f2bf(c[r]);
    }
  }
}

// ---------------------------------------------------------------------------
// Kernel 2: per-key-column softmax stats over the query axis.
// m[t] = max_{q>=t} S[q,t], l[t] = sum_{q>=t} exp(S[q,t]-m[t]), online over q.
// One block per (b, 64-column t-tile); K B-fragments held in registers.
// ---------------------------------------------------------------------------
__global__ __launch_bounds__(256) void stats_kernel(
    const __bf16* __restrict__ Qp, const __bf16* __restrict__ Kp,
    float* __restrict__ mbuf, float* __restrict__ lbuf) {
  __shared__ float sm[8][4][16][2];
  __shared__ float sl[8][4][16][2];
  const int tid = threadIdx.x;
  const int b = blockIdx.x >> 6;
  const int t0 = (blockIdx.x & 63) * 64;
  const int wave = tid >> 5, lane = tid & 31;
  const int lrow = lane & 15, lhalf = lane >> 4;

  v16bf bkf[4][2];
#pragma unroll
  for (int sub = 0; sub < 4; ++sub)
#pragma unroll
    for (int kk = 0; kk < 2; ++kk)
      bkf[sub][kk] =
          load_b_frag(Kp + ((size_t)(b * T_DIM + t0 + sub * 16)) * KP_DIM + kk * 32,
                      KP_DIM);

  float mcol[4], lcol[4];
#pragma unroll
  for (int s = 0; s < 4; ++s) { mcol[s] = NEG_BIG; lcol[s] = 0.f; }

  const int qb0 = t0 >> 4;  // q-blocks fully below the diagonal are all-masked
  for (int qb = qb0 + wave; qb < T_DIM / 16; qb += 8) {
    const __bf16* qrow = Qp + ((size_t)(b * T_DIM + qb * 16)) * KP_DIM;
    if (qb + 8 < T_DIM / 16)  // prefetch next q-block for this wave
      __builtin_prefetch(qrow + (size_t)8 * 16 * KP_DIM + (lane & 15) * KP_DIM, 0, 3);
    v16bf aq0 = load_a_frag(qrow, KP_DIM);
    v16bf aq1 = load_a_frag(qrow + 32, KP_DIM);
#pragma unroll
    for (int sub = 0; sub < 4; ++sub) {
      v8f c;
#pragma unroll
      for (int r = 0; r < 8; ++r) c[r] = 0.f;
      c = wmma_bf16(aq0, bkf[sub][0], c);
      c = wmma_bf16(aq1, bkf[sub][1], c);

      const int tcol = t0 + sub * 16 + lrow;   // this lane's column t
      const int qbase = qb * 16 + 8 * lhalf;
#pragma unroll
      for (int r = 0; r < 8; ++r)
        if (qbase + r < tcol) c[r] = NEG_BIG;  // causal: keep q >= t

      float tm = NEG_BIG;
#pragma unroll
      for (int r = 0; r < 8; ++r) tm = fmaxf(tm, c[r]);
      if (tm > 0.5f * NEG_BIG) {
        const float newm = fmaxf(mcol[sub], tm);
        float s = 0.f;
#pragma unroll
        for (int r = 0; r < 8; ++r) s += __expf(c[r] - newm);  // masked -> 0
        lcol[sub] = lcol[sub] * __expf(mcol[sub] - newm) + s;
        mcol[sub] = newm;
      }
    }
  }

#pragma unroll
  for (int sub = 0; sub < 4; ++sub) {
    sm[wave][sub][lrow][lhalf] = mcol[sub];
    sl[wave][sub][lrow][lhalf] = lcol[sub];
  }
  __syncthreads();
  if (tid < 64) {
    const int sub = tid >> 4, n = tid & 15;
    float M = NEG_BIG;
    for (int w = 0; w < 8; ++w)
      for (int h = 0; h < 2; ++h) M = fmaxf(M, sm[w][sub][n][h]);
    float L = 0.f;
    for (int w = 0; w < 8; ++w)
      for (int h = 0; h < 2; ++h) L += sl[w][sub][n][h] * __expf(sm[w][sub][n][h] - M);
    mbuf[b * T_DIM + t0 + tid] = M;
    lbuf[b * T_DIM + t0 + tid] = L;
  }
}

// ---------------------------------------------------------------------------
// Kernel 3: recompute S tiles, P = exp(S-m[t]) / (8*l[t]), read += P @ V,
// fused epilogue out[b][d][t] = x + read in the output's [B,D,T] layout.
// One wave per 16 query rows; P re-layout D-frag -> A-frag via per-wave LDS.
// ---------------------------------------------------------------------------
__global__ __launch_bounds__(256) void attend_kernel(
    const float* __restrict__ mbin,
    const __bf16* __restrict__ Qp, const __bf16* __restrict__ Kp,
    const __bf16* __restrict__ VpT,
    const float* __restrict__ mbuf, const float* __restrict__ lbuf,
    float* __restrict__ out) {
  __shared__ __bf16 pst[8][16 * 32];  // per-wave P staging [q=16][t=32]
  const int tid = threadIdx.x;
  const int b = blockIdx.x >> 5;                       // T/128 = 32 tiles
  const int wave = tid >> 5, lane = tid & 31;
  const int q0 = ((blockIdx.x & 31) * 8 + wave) * 16;  // this wave's 16 q rows
  const int lrow = lane & 15, lhalf = lane >> 4;

  const __bf16* qrow = Qp + ((size_t)(b * T_DIM + q0)) * KP_DIM;
  v16bf aq0 = load_a_frag(qrow, KP_DIM);
  v16bf aq1 = load_a_frag(qrow + 32, KP_DIM);

  v8f acc[8];
#pragma unroll
  for (int v = 0; v < 8; ++v)
#pragma unroll
    for (int r = 0; r < 8; ++r) acc[v][r] = 0.f;

  __bf16* myp = pst[wave];
  const int nblk32 = (q0 + 16 + 31) / 32;  // t <= q0+15
  for (int tb = 0; tb < nblk32; ++tb) {
    // Prefetch next block's K rows and V rows while this block computes.
    if (tb + 1 < nblk32) {
      __builtin_prefetch(
          Kp + ((size_t)(b * T_DIM + (tb + 1) * 32)) * KP_DIM + lrow * KP_DIM, 0, 3);
      __builtin_prefetch(
          VpT + ((size_t)(b * V_DIM + lrow * 8)) * T_DIM + (tb + 1) * 32, 0, 3);
    }
#pragma unroll
    for (int hb = 0; hb < 2; ++hb) {
      const int t16 = tb * 32 + hb * 16;
      if (t16 > q0 + 15) {  // entirely above diagonal: P = 0
#pragma unroll
        for (int r = 0; r < 8; ++r)
          myp[(r + 8 * lhalf) * 32 + hb * 16 + lrow] = f2bf(0.f);
      } else {
        const __bf16* krow = Kp + ((size_t)(b * T_DIM + t16)) * KP_DIM;
        v16bf bk0 = load_b_frag(krow, KP_DIM);
        v16bf bk1 = load_b_frag(krow + 32, KP_DIM);
        v8f c;
#pragma unroll
        for (int r = 0; r < 8; ++r) c[r] = 0.f;
        c = wmma_bf16(aq0, bk0, c);
        c = wmma_bf16(aq1, bk1, c);

        const int tcol = t16 + lrow;
        const float mval = mbuf[b * T_DIM + tcol];
        const float lsc = 0.125f / lbuf[b * T_DIM + tcol];  // 1/(sqrt(64)*l)
        const int qbase = q0 + 8 * lhalf;
#pragma unroll
        for (int r = 0; r < 8; ++r) {
          float p = __expf(c[r] - mval) * lsc;
          if (qbase + r < tcol) p = 0.f;  // causal
          myp[(r + 8 * lhalf) * 32 + hb * 16 + lrow] = f2bf(p);
        }
      }
    }
    // LDS in-order per wave: safe to read P back as an A-fragment.
    v16bf ap = load_a_frag(myp, 32);
#pragma unroll
    for (int vs = 0; vs < 8; ++vs) {
      v16bf bvf = load_b_frag(
          VpT + ((size_t)(b * V_DIM + vs * 16)) * T_DIM + tb * 32, T_DIM);
      acc[vs] = wmma_bf16(ap, bvf, acc[vs]);
    }
  }

  // Epilogue: out[b][v][q] = mb[b][v][q] + read[q][v]; contiguous f32 over r.
#pragma unroll
  for (int vs = 0; vs < 8; ++vs) {
    const int v = vs * 16 + lrow;
    const size_t rowbase = ((size_t)(b * V_DIM + v)) * T_DIM + q0 + 8 * lhalf;
#pragma unroll
    for (int r = 0; r < 8; ++r) out[rowbase + r] = acc[vs][r] + mbin[rowbase + r];
  }
}

// ---------------------------------------------------------------------------
extern "C" void kernel_launch(void* const* d_in, const int* in_sizes, int n_in,
                              void* d_out, int out_size, void* d_ws, size_t ws_size,
                              hipStream_t stream) {
  const float* mb = (const float*)d_in[0];
  const float* Wk = (const float*)d_in[1];
  const float* bk = (const float*)d_in[2];
  const float* Wq = (const float*)d_in[3];
  const float* bq = (const float*)d_in[4];
  const float* Wv = (const float*)d_in[5];
  const float* bv = (const float*)d_in[6];
  float* out = (float*)d_out;

  // Workspace layout (~8.7 MB):
  //   [0,2M)        Qp  bf16 [B,T,64]
  //   [2M,4M)       Kp  bf16 [B,T,64]
  //   [4M,8M)       VpT bf16 [B,128,T]
  //   [8M,8M+64K)   mbuf f32 [B,T]
  //   [.., +64K)    lbuf f32 [B,T]
  //   [.., +64K)    Wb  bf16 [256,128]  (Wq|Wk|Wv converted)
  char* ws = (char*)d_ws;
  __bf16* Qp   = (__bf16*)(ws);
  __bf16* Kp   = (__bf16*)(ws + (size_t)2 * 1024 * 1024);
  __bf16* VpT  = (__bf16*)(ws + (size_t)4 * 1024 * 1024);
  float*  mbuf = (float*)(ws + (size_t)8 * 1024 * 1024);
  float*  lbuf = (float*)(ws + (size_t)8 * 1024 * 1024 + 64 * 1024);
  __bf16* Wb   = (__bf16*)(ws + (size_t)8 * 1024 * 1024 + 128 * 1024);

  hipLaunchKernelGGL(wconv_kernel, dim3(32), dim3(256), 0, stream, Wq, Wk, Wv, Wb);
  hipLaunchKernelGGL(proj_kernel, dim3(B_DIM * (T_DIM / 64)), dim3(256), 0, stream,
                     mb, Wb, bk, bq, bv, Qp, Kp, VpT);
  hipLaunchKernelGGL(stats_kernel, dim3(B_DIM * (T_DIM / 64)), dim3(256), 0, stream,
                     Qp, Kp, mbuf, lbuf);
  hipLaunchKernelGGL(attend_kernel, dim3(B_DIM * (T_DIM / 128)), dim3(256), 0, stream,
                     mb, Qp, Kp, VpT, mbuf, lbuf, out);
}
